// VectorQuantizer_3204045603082
// MI455X (gfx1250) — compile-verified, matrix operand order checked by reading, not run on hardware
//
#include <hip/hip_runtime.h>
#include <hip/hip_bf16.h>
#include <stdint.h>

typedef __attribute__((ext_vector_type(16))) __bf16        v16bf;
typedef __attribute__((ext_vector_type(16))) unsigned short v16us;
typedef __attribute__((ext_vector_type(8)))  float          v8f;
typedef __attribute__((ext_vector_type(8)))  unsigned int   v8u;

#define N_TOTAL  65536          // B*D*H*W
#define CDIM     256            // embedding dim
#define KCODES   1024           // codebook size
#define SSPAT    16384          // D*H*W (spatial stride between channels)
#define NKCHUNK  64             // 1024 / 16
#define NCBLK    8              // 256 / 32
#define ZQ_ELEMS 16777216       // 4*256*16*32*32

// workspace byte offsets
#define WS_EMBB  0              // packed B tiles: 64*8*32*8 dwords = 524288 B
#define WS_ESQ   524288         // 1024 floats
#define WS_IDX   528384         // 65536 int32
#define WS_ACC   790528         // 1 float accumulator

__device__ __forceinline__ uint16_t f32_to_bf16_rne(float f) {
    uint32_t u = __float_as_uint(f);
    return (uint16_t)((u + 0x7FFFu + ((u >> 16) & 1u)) >> 16);
}

// ---------------------------------------------------------------------------
// K0: e_sq, pack codebook into WMMA-B layout (bf16), zero loss accumulator.
// One block per code k; thread = channel c.
// B-operand layout (16-bit, 32x16, mirror of A table in ISA 7.12.2):
//   lane = (k&15) + 16*((cc>>3)&1); vgpr = ((cc>>4)<<2) + ((cc&7)>>1); half = cc&1
// ---------------------------------------------------------------------------
__global__ void vq_prep(const float* __restrict__ emb, uint16_t* __restrict__ embB,
                        float* __restrict__ esq, float* __restrict__ acc) {
    const int k = blockIdx.x;
    const int c = threadIdx.x;
    const float v = emb[k * CDIM + c];

    const int cc   = c & 31;
    const int cb   = c >> 5;
    const int j    = k >> 4;
    const int col  = k & 15;
    const int half = (cc >> 3) & 1;
    const int lane = col + 16 * half;
    const int vgpr = ((cc >> 4) << 2) + ((cc & 7) >> 1);
    const int dw   = ((j * NCBLK + cb) * 32 + lane) * 8 + vgpr;
    embB[dw * 2 + (c & 1)] = f32_to_bf16_rne(v);

    __shared__ float red[256];
    red[c] = v * v;
    __syncthreads();
    for (int s = 128; s > 0; s >>= 1) {
        if (c < s) red[c] += red[c + s];
        __syncthreads();
    }
    if (c == 0) esq[k] = red[0];
    if (k == 0 && c == 0) *acc = 0.0f;
}

// ---------------------------------------------------------------------------
// K1: distance GEMM via v_wmma_f32_16x16x32_bf16 + per-row argmin.
// Block = 256 threads = 8 waves; each wave owns a 16-row n-tile (block = 128 n).
// argmin_k(||z||^2 + ||e||^2 - 2 z.e) == argmin_k(||e||^2 - 2 z.e)  -> no z_sq.
// ---------------------------------------------------------------------------
__global__ void __launch_bounds__(256)
vq_argmin(const float* __restrict__ z, const uint32_t* __restrict__ embB,
          const float* __restrict__ esq, int* __restrict__ idx_ws,
          int* __restrict__ idx_out) {
    __shared__ float s_esq[KCODES];
    for (int i = threadIdx.x; i < KCODES; i += 256) s_esq[i] = esq[i];
    __syncthreads();

    const int lane = threadIdx.x & 31;
    const int wave = threadIdx.x >> 5;
    const int n0   = blockIdx.x * 128 + wave * 16;
    const int half = lane >> 4;           // which K-half this lane packs
    const int m    = lane & 15;           // A row / k column within tile
    const int n    = n0 + m;              // n-row this lane loads A data for
    const size_t zbase =
        (size_t)(n >> 14) * CDIM * SSPAT + (size_t)(n & (SSPAT - 1));

    // Load + pack A tile once: 8 c-blocks of 32 channels, bf16, A-operand layout.
    v16bf A[NCBLK];
#pragma unroll
    for (int cb = 0; cb < NCBLK; ++cb) {
        v16us au;
#pragma unroll
        for (int i = 0; i < 16; ++i) {
            const int cc = (i < 8) ? (half * 8 + i) : (16 + half * 8 + (i - 8));
            const int c  = cb * 32 + cc;
            au[i] = f32_to_bf16_rne(z[zbase + (size_t)c * SSPAT]);
        }
        A[cb] = __builtin_bit_cast(v16bf, au);
    }

    float minv[8];
    int   mink[8];
#pragma unroll
    for (int r = 0; r < 8; ++r) { minv[r] = 3.4e38f; mink[r] = 0; }

    for (int j = 0; j < NKCHUNK; ++j) {
        v8f accd = {};
        const uint32_t* bbase = embB + (size_t)j * NCBLK * 32 * 8;
#pragma unroll
        for (int cb = 0; cb < NCBLK; ++cb) {
            v8u   braw = *(const v8u*)(bbase + (cb * 32 + lane) * 8);
            v16bf Bm   = __builtin_bit_cast(v16bf, braw);
            accd = __builtin_amdgcn_wmma_f32_16x16x32_bf16(
                false, A[cb], false, Bm, (short)0, accd, false, false);
        }
        const int   kk = j * 16 + m;      // k column held by this lane
        const float ek = s_esq[kk];
#pragma unroll
        for (int r = 0; r < 8; ++r) {
            const float d = fmaf(-2.0f, accd[r], ek);
            if (d < minv[r]) { minv[r] = d; mink[r] = kk; }
        }
    }

    // Reduce across the 16 k-lanes of each half (xor masks 1..8 stay in-half).
#pragma unroll
    for (int r = 0; r < 8; ++r) {
        float v = minv[r];
        int   k = mink[r];
#pragma unroll
        for (int off = 1; off < 16; off <<= 1) {
            const float ov = __shfl_xor(v, off, 32);
            const int   ok = __shfl_xor(k, off, 32);
            if (ov < v || (ov == v && ok < k)) { v = ov; k = ok; }
        }
        if (m == 0) {
            const int nn = n0 + half * 8 + r;  // C/D layout: row = r + 8*half
            idx_ws[nn]  = k;
            idx_out[nn] = k;
        }
    }
}

// ---------------------------------------------------------------------------
// K2: z_q = emb[idx], write in (B,C,D,H,W) layout; accumulate sum((z_q-z)^2).
// grid.x = n-chunks of 64 (coalesced: consecutive n = consecutive addresses),
// grid.y = c-chunks of 4.
// ---------------------------------------------------------------------------
__global__ void vq_gather(const float* __restrict__ z, const float* __restrict__ emb,
                          const int* __restrict__ idx_ws, float* __restrict__ zq_out,
                          float* __restrict__ acc) {
    const int nl = threadIdx.x & 63;
    const int cl = threadIdx.x >> 6;
    const int n  = blockIdx.x * 64 + nl;
    const int c  = blockIdx.y * 4 + cl;
    const int k  = idx_ws[n];
    const float q = emb[k * CDIM + c];
    const size_t a =
        ((size_t)(n >> 14) * CDIM + (size_t)c) * SSPAT + (size_t)(n & (SSPAT - 1));
    const float d = q - z[a];
    zq_out[a] = q;                         // straight-through value == z_q

    __shared__ float red[256];
    red[threadIdx.x] = d * d;
    __syncthreads();
    for (int s = 128; s > 0; s >>= 1) {
        if (threadIdx.x < s) red[threadIdx.x] += red[threadIdx.x + s];
        __syncthreads();
    }
    if (threadIdx.x == 0) atomicAdd(acc, red[0]);
}

// K3: loss = (1 + BETA) * mean(diff^2), BETA = 0.25
__global__ void vq_loss(const float* __restrict__ acc, float* __restrict__ out_loss) {
    *out_loss = 1.25f * (*acc) / 16777216.0f;
}

// ---------------------------------------------------------------------------
extern "C" void kernel_launch(void* const* d_in, const int* in_sizes, int n_in,
                              void* d_out, int out_size, void* d_ws, size_t ws_size,
                              hipStream_t stream) {
    const float* z   = (const float*)d_in[0];  // (4,256,16,32,32) f32
    const float* emb = (const float*)d_in[1];  // (1024,256) f32

    uint8_t*  ws     = (uint8_t*)d_ws;
    uint16_t* embB16 = (uint16_t*)(ws + WS_EMBB);
    uint32_t* embB32 = (uint32_t*)(ws + WS_EMBB);
    float*    esq    = (float*)(ws + WS_ESQ);
    int*      idx_ws = (int*)(ws + WS_IDX);
    float*    acc    = (float*)(ws + WS_ACC);

    float* out     = (float*)d_out;
    float* zq      = out;                        // 16777216 f32
    float* loss    = out + ZQ_ELEMS;             // 1 f32
    int*   idx_out = (int*)(out + ZQ_ELEMS + 1); // 65536 int32 (raw bits)

    vq_prep<<<KCODES, 256, 0, stream>>>(emb, embB16, esq, acc);
    vq_argmin<<<N_TOTAL / 128, 256, 0, stream>>>(z, embB32, esq, idx_ws, idx_out);
    dim3 g(N_TOTAL / 64, CDIM / 4);
    vq_gather<<<g, 256, 0, stream>>>(z, emb, idx_ws, zq, acc);
    vq_loss<<<1, 1, 0, stream>>>(acc, loss);
}
